// SpectralConvCross1d_local_51857435131962
// MI455X (gfx1250) — compile-verified, matrix operand order checked by reading, not run on hardware
//
// SpectralConvCross1d — MI455X (gfx1250) implementation.
// Local attention (flash, f16 WMMA) + truncated-mode DFT cross attention
// (DFT/iDFT expressed as f16 WMMA GEMMs against precomputed bases).
// Round 3: fix async-builtin pointer types (v4i32 pointers in AS1/AS3 per
//          the clang diagnostic), keep async LDS staging + prefetch.
#include <hip/hip_runtime.h>
#include <hip/hip_bf16.h>

typedef __attribute__((ext_vector_type(16))) _Float16 v16h;
typedef __attribute__((ext_vector_type(8)))  float    v8f;

#define LQ 1024   // sequence length
#define NB 8      // batch
#define NH 8      // heads
#define NE 64     // head dim == in_ch/8 == out_ch/8
#define NM 64     // retained modes
#define RI 128    // re||im concatenated width

// ---- CDNA5 async global->LDS copy (GLOBAL_LOAD_ASYNC_TO_LDS_B128, ASYNCcnt) ---------
#if __has_builtin(__builtin_amdgcn_global_load_async_to_lds_b128) && \
    __has_builtin(__builtin_amdgcn_s_wait_asynccnt)
#define CDNA5_ASYNC_LDS 1
#else
#define CDNA5_ASYNC_LDS 0
#endif

typedef int async_v4i __attribute__((vector_size(16)));

static __device__ __forceinline__ void copy16_g2l(const void* g, void* l) {
#if CDNA5_ASYNC_LDS
  __builtin_amdgcn_global_load_async_to_lds_b128(
      (__attribute__((address_space(1))) async_v4i*)g,
      (__attribute__((address_space(3))) async_v4i*)l, 0, 0);
#else
  *(uint4*)l = *(const uint4*)g;
#endif
}

static __device__ __forceinline__ void copy16_wait() {
#if CDNA5_ASYNC_LDS
  __builtin_amdgcn_s_wait_asynccnt(0);
#endif
}

// ---- WMMA fragment helpers (gfx1250 wave32 16-bit layouts, CDNA5 ISA) ----

// K-offset of packed pair p (0..7) within a 32-wide K window.
static __device__ __forceinline__ int frag_koff(int p, int lane) {
  return ((p < 4) ? (2 * p) : (2 * p + 8)) + ((lane & 16) ? 8 : 0);
}

// Fragment where element(i, k) = base[i*ld + k], i = lane&15 (A row-major, or B from an NxK array).
static __device__ __forceinline__ v16h frag_rowmajor(const _Float16* base, int ld) {
  const int lane = (int)(threadIdx.x & 31u);
  const _Float16* rp = base + (lane & 15) * ld;
  v16h f;
#pragma unroll
  for (int p = 0; p < 8; ++p) {
    const int k = frag_koff(p, lane);
    f[2 * p]     = rp[k];
    f[2 * p + 1] = rp[k + 1];
  }
  return f;
}

// Fragment where element(i, k) = base[k*ld + i] (A stored K-major, or B from a KxN array).
static __device__ __forceinline__ v16h frag_colmajor(const _Float16* base, int ld) {
  const int lane = (int)(threadIdx.x & 31u);
  const _Float16* cp = base + (lane & 15);
  v16h f;
#pragma unroll
  for (int p = 0; p < 8; ++p) {
    const int k = frag_koff(p, lane);
    f[2 * p]     = cp[k * ld];
    f[2 * p + 1] = cp[(k + 1) * ld];
  }
  return f;
}

static __device__ __forceinline__ v8f wmma16(v16h a, v16h b, v8f c) {
  // D = A(16x32 f16) * B(32x16 f16) + C(16x16 f32)
  return __builtin_amdgcn_wmma_f32_16x16x32_f16(false, a, false, b, (short)0, c, false, false);
}

static __device__ __forceinline__ v8f vzero8() {
  v8f z;
#pragma unroll
  for (int i = 0; i < 8; ++i) z[i] = 0.0f;
  return z;
}

// ---- Kernel 0: DFT bases ------------------------------------------------------------
// fwd[l][m]      = cos(2*pi*l*m/1024),  fwd[l][64+m] = -sin(...)        (1024 x 128)
// inv[m][l]      = c_m*cos(2*pi*m*l/1024), inv[64+m][l] = -c_m*sin(...) (128 x 1024), c_0=1 else 2
__global__ __launch_bounds__(256) void basis_kernel(_Float16* __restrict__ fwd,
                                                    _Float16* __restrict__ inv) {
  const int idx = (int)(blockIdx.x * 256 + threadIdx.x);
  const float w0 = 6.283185307179586f / 1024.0f;
  if (idx < LQ * NM) {
    const int l = idx >> 6, m = idx & 63;
    const int ph = (l * m) & 1023;
    float sn, cs;
    __sincosf(w0 * (float)ph, &sn, &cs);
    fwd[l * RI + m]      = (_Float16)cs;
    fwd[l * RI + 64 + m] = (_Float16)(-sn);
  }
  if (idx < RI * LQ) {
    const int k = idx >> 10, l = idx & 1023;
    const int m = k & 63;
    const float coef = (m == 0) ? 1.0f : 2.0f;
    const int ph = (l * m) & 1023;
    float sn, cs;
    __sincosf(w0 * (float)ph, &sn, &cs);
    inv[k * LQ + l] = (_Float16)((k < 64) ? coef * cs : -coef * sn);
  }
}

// ---- Kernel 1: flash attention, writes 0.5*V_local into d_out -----------------------
#define LDA 72  // LDS leading dim (halfs): stride 36 banks -> conflict-free strided frags

__global__ __launch_bounds__(128) void attn_kernel(const float* __restrict__ q,
                                                   const float* __restrict__ k,
                                                   const float* __restrict__ v,
                                                   float* __restrict__ out) {
  __shared__ _Float16 Qs[64 * LDA];
  __shared__ _Float16 Ks[64 * LDA];
  __shared__ _Float16 Vs[64 * LDA];
  __shared__ _Float16 Ps[4 * 16 * LDA];

  const int qt = (int)blockIdx.x;  // query tile 0..15
  const int h  = (int)blockIdx.y;
  const int b  = (int)blockIdx.z;
  const int tid  = (int)threadIdx.x;
  const int wave = tid >> 5;
  const int lane = tid & 31;
  const int m0   = wave * 16;
  const float scale = 1.0f / (512.0f * 512.0f);

  {  // stage Q tile (64 rows x 64 e) as f16
    const float* qb = q + (((size_t)b * LQ + (size_t)qt * 64) * NH + h) * NE;
#pragma unroll
    for (int i = 0; i < 8; ++i) {
      const int idx = tid + i * 128;         // 1024 float4s
      const int row = idx >> 4, c4 = (idx & 15) * 4;
      const float4 f = *(const float4*)(qb + (size_t)row * NH * NE + c4);
      _Float16* d = Qs + row * LDA + c4;
      d[0] = (_Float16)f.x; d[1] = (_Float16)f.y; d[2] = (_Float16)f.z; d[3] = (_Float16)f.w;
    }
  }

  float mrun[8], lrun[8];
  v8f oacc[4];
#pragma unroll
  for (int r = 0; r < 8; ++r) { mrun[r] = -1e30f; lrun[r] = 0.0f; }
#pragma unroll
  for (int n = 0; n < 4; ++n) oacc[n] = vzero8();

  for (int kt = 0; kt < 16; ++kt) {
    __syncthreads();  // protect previous-iteration K/V reads
    const float* kb = k + (((size_t)b * LQ + (size_t)kt * 64) * NH + h) * NE;
    const float* vb = v + (((size_t)b * LQ + (size_t)kt * 64) * NH + h) * NE;
#pragma unroll
    for (int i = 0; i < 8; ++i) {
      const int idx = tid + i * 128;
      const int row = idx >> 4, c4 = (idx & 15) * 4;
      const float4 fk = *(const float4*)(kb + (size_t)row * NH * NE + c4);
      const float4 fv = *(const float4*)(vb + (size_t)row * NH * NE + c4);
      _Float16* dk = Ks + row * LDA + c4;
      dk[0] = (_Float16)fk.x; dk[1] = (_Float16)fk.y; dk[2] = (_Float16)fk.z; dk[3] = (_Float16)fk.w;
      _Float16* dv = Vs + row * LDA + c4;
      dv[0] = (_Float16)fv.x; dv[1] = (_Float16)fv.y; dv[2] = (_Float16)fv.z; dv[3] = (_Float16)fv.w;
    }
    // prefetch next iteration's K/V rows into cache (global_prefetch_b8)
    if (kt < 15) {
      const size_t prow = (size_t)(tid >> 1) * NH * NE + (size_t)(tid & 1) * 32;
      __builtin_prefetch(kb + (size_t)64 * NH * NE + prow, 0, 0);
      __builtin_prefetch(vb + (size_t)64 * NH * NE + prow, 0, 0);
    }
    __syncthreads();

    // S strip (16x64) = Q[m0..m0+15,:] * K^T  — K^T fragment == row-major frag on K rows
    const v16h aq0 = frag_rowmajor(Qs + m0 * LDA, LDA);
    const v16h aq1 = frag_rowmajor(Qs + m0 * LDA + 32, LDA);
    v8f s[4];
#pragma unroll
    for (int n = 0; n < 4; ++n) {
      v8f acc = vzero8();
      acc = wmma16(aq0, frag_rowmajor(Ks + (n * 16) * LDA, LDA), acc);
      acc = wmma16(aq1, frag_rowmajor(Ks + (n * 16) * LDA + 32, LDA), acc);
      s[n] = acc;
    }

    // online softmax: lanes 0-15 own row r, lanes 16-31 own row r+8 (C layout);
    // xor masks <=8 keep the two half-groups independent on wave32.
    float alpha[8];
#pragma unroll
    for (int r = 0; r < 8; ++r) {
      float mx = fmaxf(fmaxf(s[0][r], s[1][r]), fmaxf(s[2][r], s[3][r]));
#pragma unroll
      for (int off = 8; off >= 1; off >>= 1) mx = fmaxf(mx, __shfl_xor(mx, off, 32));
      mx *= scale;
      const float mnew = fmaxf(mrun[r], mx);
      alpha[r] = __expf(mrun[r] - mnew);
      mrun[r]  = mnew;
    }
    float rsum[8];
#pragma unroll
    for (int r = 0; r < 8; ++r) rsum[r] = 0.0f;
#pragma unroll
    for (int n = 0; n < 4; ++n) {
#pragma unroll
      for (int r = 0; r < 8; ++r) {
        const float p = __expf(scale * s[n][r] - mrun[r]);
        s[n][r] = p;
        rsum[r] += p;
      }
    }
#pragma unroll
    for (int r = 0; r < 8; ++r) {
#pragma unroll
      for (int off = 8; off >= 1; off >>= 1) rsum[r] += __shfl_xor(rsum[r], off, 32);
      lrun[r] = lrun[r] * alpha[r] + rsum[r];
    }

    // rescale O, stage P (C-layout scatter) into this wave's private LDS strip
    const int crow = (lane & 16) ? 8 : 0;
    const int ccol = lane & 15;
    _Float16* Pw = Ps + wave * 16 * LDA;
#pragma unroll
    for (int n = 0; n < 4; ++n) {
#pragma unroll
      for (int r = 0; r < 8; ++r) {
        oacc[n][r] *= alpha[r];
        Pw[(crow + r) * LDA + n * 16 + ccol] = (_Float16)s[n][r];
      }
    }

    // O(16x64) += P(16x64) * V(64x64)
    const v16h ap0 = frag_rowmajor(Pw, LDA);
    const v16h ap1 = frag_rowmajor(Pw + 32, LDA);
#pragma unroll
    for (int n = 0; n < 4; ++n) {
      oacc[n] = wmma16(ap0, frag_colmajor(Vs + n * 16, LDA), oacc[n]);
      oacc[n] = wmma16(ap1, frag_colmajor(Vs + 32 * LDA + n * 16, LDA), oacc[n]);
    }
  }

  // epilogue: d_out = 0.5 * O / l   at [b, l, h, e]
  const int crow = (lane & 16) ? 8 : 0;
  const int ccol = lane & 15;
#pragma unroll
  for (int n = 0; n < 4; ++n) {
#pragma unroll
    for (int r = 0; r < 8; ++r) {
      const int row = qt * 64 + m0 + crow + r;
      const int col = n * 16 + ccol;
      out[(((size_t)b * LQ + row) * NH + h) * NE + col] = 0.5f * oacc[n][r] / lrun[r];
    }
  }
}

// ---- Kernel 2: truncated-mode DFT as GEMM: x^T(64x1024) * basis(1024x128) ----------
#define LDX 72

__global__ __launch_bounds__(128) void dft_kernel(const float* __restrict__ q,
                                                  const float* __restrict__ k,
                                                  const _Float16* __restrict__ basis,
                                                  float* __restrict__ xqft,
                                                  float* __restrict__ xkft) {
  __shared__ _Float16 Xs[32 * LDX];   // x chunk: 32 l-rows x 64 e
  __shared__ _Float16 Bs[32 * RI];    // basis chunk: 32 l-rows x 128
  const int which = (int)blockIdx.x;  // 0=q, 1=k
  const int h = (int)blockIdx.y, b = (int)blockIdx.z;
  const int tid = (int)threadIdx.x, wave = tid >> 5, lane = tid & 31, m0 = wave * 16;
  const float* x = which ? k : q;
  float* dst = which ? xkft : xqft;

  v8f acc[8];
#pragma unroll
  for (int n = 0; n < 8; ++n) acc[n] = vzero8();

  for (int c = 0; c < 32; ++c) {
    __syncthreads();
    const int l0 = c * 32;
    // basis chunk: pure f16 copy -> async global->LDS (no VGPR round trip)
#pragma unroll
    for (int i = 0; i < 4; ++i) {
      const int idx = tid + i * 128;          // 512 b128 chunks
      const int row = idx >> 4, c8 = (idx & 15) * 8;
      copy16_g2l(basis + (size_t)(l0 + row) * RI + c8, Bs + row * RI + c8);
    }
    // x chunk needs f32->f16 conversion: VGPR path
    const float* xb = x + (((size_t)b * LQ + l0) * NH + h) * NE;
#pragma unroll
    for (int i = 0; i < 4; ++i) {
      const int idx = tid + i * 128;          // 512 float4s
      const int row = idx >> 4, c4 = (idx & 15) * 4;
      const float4 f = *(const float4*)(xb + (size_t)row * NH * NE + c4);
      _Float16* d = Xs + row * LDX + c4;
      d[0] = (_Float16)f.x; d[1] = (_Float16)f.y; d[2] = (_Float16)f.z; d[3] = (_Float16)f.w;
    }
    copy16_wait();
    __syncthreads();

    const v16h a = frag_colmajor(Xs + m0, LDX);  // element(e, l') = Xs[l'][m0+e]
#pragma unroll
    for (int n = 0; n < 8; ++n)
      acc[n] = wmma16(a, frag_colmajor(Bs + n * 16, RI), acc[n]);
  }

  float* o = dst + (size_t)(b * NH + h) * NE * RI;
  const int crow = (lane & 16) ? 8 : 0, ccol = lane & 15;
#pragma unroll
  for (int n = 0; n < 8; ++n)
#pragma unroll
    for (int r = 0; r < 8; ++r)
      o[(m0 + crow + r) * RI + n * 16 + ccol] = acc[n][r];
}

// ---- Kernel 3: complex mode-space contractions + complex tanh gate ------------------
__global__ __launch_bounds__(256) void cross_kernel(const float* __restrict__ xqft,
                                                    const float* __restrict__ xkft,
                                                    const float* __restrict__ wr,
                                                    const float* __restrict__ wi,
                                                    _Float16* __restrict__ aw) {
  __shared__ float Ash[64 * RI];  // xq -> xqk -> xqkv
  __shared__ float Ksh[64 * RI];  // xk
  const int h = (int)blockIdx.x, b = (int)blockIdx.y;
  const int tid = (int)threadIdx.x;
  const float* qsrc = xqft + (size_t)(b * NH + h) * NE * RI;
  const float* ksrc = xkft + (size_t)(b * NH + h) * NE * RI;
  // stage mode coefficients: f32, no conversion -> async global->LDS
  for (int c = tid; c < (64 * RI) / 4; c += 256) {
    copy16_g2l(qsrc + c * 4, Ash + c * 4);
    copy16_g2l(ksrc + c * 4, Ksh + c * 4);
  }
  copy16_wait();
  __syncthreads();

  float rre[16], rim[16];
  {  // xqk[x,y] = tanh_c( sum_e xq[e,x] * xk[e,y] )
    const int y = tid & 63, xbase = (tid >> 6) * 16;
#pragma unroll
    for (int i = 0; i < 16; ++i) {
      const int x = xbase + i;
      float sr = 0.0f, si = 0.0f;
      for (int e = 0; e < 64; ++e) {
        const float qr = Ash[e * RI + x], qi = Ash[e * RI + 64 + x];
        const float kr = Ksh[e * RI + y], ki = Ksh[e * RI + 64 + y];
        sr += qr * kr - qi * ki;
        si += qr * ki + qi * kr;
      }
      const float tx = tanhf(sr), ty = tanf(si);
      const float d = 1.0f + tx * tx * ty * ty;
      rre[i] = tx * (1.0f + ty * ty) / d;
      rim[i] = ty * (1.0f - tx * tx) / d;
    }
  }
  __syncthreads();
  {  // overwrite Ash with xqk as [x][y re | y im]
    const int y = tid & 63, xbase = (tid >> 6) * 16;
#pragma unroll
    for (int i = 0; i < 16; ++i) {
      Ash[(xbase + i) * RI + y]      = rre[i];
      Ash[(xbase + i) * RI + 64 + y] = rim[i];
    }
  }
  __syncthreads();
  {  // xqkv[e,x] = sum_y xqk[x,y] * xk[e,y]
    const int x = tid & 63, ebase = (tid >> 6) * 16;
#pragma unroll
    for (int i = 0; i < 16; ++i) {
      const int e = ebase + i;
      float sr = 0.0f, si = 0.0f;
      for (int y = 0; y < 64; ++y) {
        const float ar = Ash[x * RI + y], ai = Ash[x * RI + 64 + y];
        const float kr = Ksh[e * RI + y], ki = Ksh[e * RI + 64 + y];
        sr += ar * kr - ai * ki;
        si += ar * ki + ai * kr;
      }
      rre[i] = sr; rim[i] = si;
    }
  }
  __syncthreads();
  {  // overwrite Ash with xqkv as [e][x re | x im]
    const int x = tid & 63, ebase = (tid >> 6) * 16;
#pragma unroll
    for (int i = 0; i < 16; ++i) {
      Ash[(ebase + i) * RI + x]      = rre[i];
      Ash[(ebase + i) * RI + 64 + x] = rim[i];
    }
  }
  __syncthreads();
  {  // xqkvw[o,x] = sum_e xqkv[e,x] * (wr + i*wi)[h,e,o,x] -> f16, A-fragment-ready
    const int x = tid & 63, obase = (tid >> 6) * 16;
    _Float16* dst = aw + (size_t)(b * NH + h) * NE * RI;
#pragma unroll
    for (int i = 0; i < 16; ++i) {
      const int o = obase + i;
      float sr = 0.0f, si = 0.0f;
      for (int e = 0; e < 64; ++e) {
        const float ar = Ash[e * RI + x], ai = Ash[e * RI + 64 + x];
        const size_t wofs = (((size_t)h * NE + e) * NE + o) * NM + x;
        const float wre = wr[wofs], wim = wi[wofs];
        sr += ar * wre - ai * wim;
        si += ar * wim + ai * wre;
      }
      dst[o * RI + x]      = (_Float16)sr;
      dst[o * RI + 64 + x] = (_Float16)si;
    }
  }
}

// ---- Kernel 4: iDFT as GEMM: A(64x128) * inv(128x1024), fused add into d_out --------
#define LDAI 136  // 272B row stride -> 16-lane strided frag reads hit distinct banks

__global__ __launch_bounds__(128) void idft_kernel(const _Float16* __restrict__ aw,
                                                   const _Float16* __restrict__ ibasis,
                                                   float* __restrict__ out) {
  __shared__ _Float16 As[64 * LDAI];
  __shared__ _Float16 Bs[128 * 64];
  const int h = (int)blockIdx.x, b = (int)blockIdx.y;
  const int tid = (int)threadIdx.x, wave = tid >> 5, lane = tid & 31, m0 = wave * 16;

  const _Float16* asrc = aw + (size_t)(b * NH + h) * NE * RI;
  for (int idx = tid; idx < 64 * 16; idx += 128) {  // 64 rows * 16 b128 chunks (async)
    const int row = idx >> 4, c8 = (idx & 15) * 8;
    copy16_g2l(asrc + (size_t)row * RI + c8, As + row * LDAI + c8);
  }
  copy16_wait();
  __syncthreads();

  v16h af[4];
#pragma unroll
  for (int kk = 0; kk < 4; ++kk) af[kk] = frag_rowmajor(As + m0 * LDAI + kk * 32, LDAI);

  const float s = 0.5f / (1024.0f * 512.0f * 512.0f);
  const int crow = (lane & 16) ? 8 : 0, ccol = lane & 15;

  for (int ns = 0; ns < 16; ++ns) {
    __syncthreads();
    const int n0 = ns * 64;
    for (int idx = tid; idx < 128 * 8; idx += 128) {  // 128 rows * 8 b128 chunks (async)
      const int row = idx >> 3, c8 = (idx & 7) * 8;
      copy16_g2l(ibasis + (size_t)row * LQ + n0 + c8, Bs + row * 64 + c8);
    }
    copy16_wait();
    __syncthreads();
#pragma unroll
    for (int n = 0; n < 4; ++n) {
      v8f acc = vzero8();
#pragma unroll
      for (int kk = 0; kk < 4; ++kk)
        acc = wmma16(af[kk], frag_colmajor(Bs + kk * 32 * 64 + n * 16, 64), acc);
#pragma unroll
      for (int r = 0; r < 8; ++r) {
        const int o = m0 + crow + r;
        const int l = n0 + n * 16 + ccol;
        const size_t oi = (((size_t)b * LQ + l) * NH + h) * NE + o;
        out[oi] += s * acc[r];
      }
    }
  }
}

// ---- host launch --------------------------------------------------------------------
extern "C" void kernel_launch(void* const* d_in, const int* in_sizes, int n_in,
                              void* d_out, int out_size, void* d_ws, size_t ws_size,
                              hipStream_t stream) {
  (void)in_sizes; (void)n_in; (void)out_size; (void)ws_size;
  const float* q  = (const float*)d_in[0];
  const float* k  = (const float*)d_in[1];
  const float* v  = (const float*)d_in[2];
  // d_in[3] = mask (unused: non-None mask skips masking), d_in[6]/[7] = identity index gathers
  const float* wr = (const float*)d_in[4];
  const float* wi = (const float*)d_in[5];
  float* out = (float*)d_out;

  char* ws = (char*)d_ws;
  _Float16* fwd  = (_Float16*)(ws);                              // 1024*128*2 = 256 KB
  _Float16* inv  = (_Float16*)(ws + (256 << 10));                // 128*1024*2 = 256 KB
  float*    xqft = (float*)(ws + (512 << 10));                   // 8*8*64*128*4 = 2 MB
  float*    xkft = (float*)(ws + (512 << 10) + (2 << 20));       // 2 MB
  _Float16* aw   = (_Float16*)(ws + (512 << 10) + (4 << 20));    // 1 MB

  basis_kernel<<<512, 256, 0, stream>>>(fwd, inv);
  attn_kernel<<<dim3(16, NH, NB), 128, 0, stream>>>(q, k, v, out);
  dft_kernel<<<dim3(2, NH, NB), 128, 0, stream>>>(q, k, fwd, xqft, xkft);
  cross_kernel<<<dim3(NH, NB), 256, 0, stream>>>(xqft, xkft, wr, wi, aw);
  idft_kernel<<<dim3(NH, NB), 128, 0, stream>>>(aw, inv, out);
}
